// ViTAttentionBlock_7550552506788
// MI455X (gfx1250) — matrix operand order repaired and checked by reading, not compiled, not run on hardware
//
#include <hip/hip_runtime.h>
#include <hip/hip_bf16.h>

typedef __bf16 bf16_t;
typedef __attribute__((ext_vector_type(16))) __bf16 v16bf;
typedef __attribute__((ext_vector_type(8)))  __bf16 v8bf;
typedef __attribute__((ext_vector_type(8)))  float   v8f;

#define EMBED       768
#define THREE_EMBED 2304
#define HEADS       12
#define HDIM        64
#define BATCH       8
#define SEQ         1024
#define TOKENS      (BATCH * SEQ)   // 8192

union FragU { v16bf v; v8bf h[2]; };

__device__ __forceinline__ v8f wmma_bf16f32(v16bf a, v16bf b, v8f c) {
  // D = A(16x32 bf16) * B(32x16 bf16) + C(16x16 f32)
  return __builtin_amdgcn_wmma_f32_16x16x32_bf16(
      /*neg_a=*/false, a, /*neg_b=*/false, b,
      /*c_mod=*/(short)0, c, /*reuse_a=*/false, /*reuse_b=*/false);
}

// A-matrix fragment (16x32, bf16), row-major source with leading dim `ld`.
// lane L<16: row L, K = k0+{0..7, 16..23}; lane L+16: row L, K = k0+{8..15, 24..31}
__device__ __forceinline__ v16bf load_frag_a(const bf16_t* base, int ld, int lane, int k0) {
  const int row = lane & 15;
  const int kb  = k0 + ((lane >> 4) << 3);
  const bf16_t* p = base + (size_t)row * ld + kb;
  FragU u;
  u.h[0] = *(const v8bf*)(p);
  u.h[1] = *(const v8bf*)(p + 16);
  return u.v;
}

// B-matrix fragment (32x16, bf16) from TRANSPOSED storage: base[col][k], leading dim `ld`.
// lane c<16: col c, K = k0+{0..7, 16..23}; lane c+16: col c, K = k0+{8..15, 24..31}
__device__ __forceinline__ v16bf load_frag_b(const bf16_t* base, int ld, int lane, int k0) {
  const int col = lane & 15;
  const int kb  = k0 + ((lane >> 4) << 3);
  const bf16_t* p = base + (size_t)col * ld + kb;
  FragU u;
  u.h[0] = *(const v8bf*)(p);
  u.h[1] = *(const v8bf*)(p + 16);
  return u.v;
}

// ---------------- conversion kernels ----------------
__global__ void __launch_bounds__(256) vit_cvt_bf16(const float* __restrict__ in,
                                                    bf16_t* __restrict__ out, int n) {
  for (int i = blockIdx.x * 256 + threadIdx.x; i < n; i += gridDim.x * 256)
    out[i] = (bf16_t)in[i];
}

// in [rows][cols] fp32 -> out [cols][rows] bf16
__global__ void __launch_bounds__(256) vit_cvt_transpose(const float* __restrict__ in,
                                                         bf16_t* __restrict__ out,
                                                         int rows, int cols) {
  const int total = rows * cols;
  for (int i = blockIdx.x * 256 + threadIdx.x; i < total; i += gridDim.x * 256) {
    const int c = i / rows;
    const int r = i - c * rows;
    out[i] = (bf16_t)in[(size_t)r * cols + c];
  }
}

// ---------------- QKV GEMM: [8192,768] @ [768,2304] ----------------
// Writes Q (scaled by d^-0.5), K as [b,h,n,d] bf16 and V transposed [b,h,d,n] bf16.
__device__ __forceinline__ void store_qkv_tile(const v8f& c, int m0, int n0, int lane,
                                               bf16_t* __restrict__ Qd,
                                               bf16_t* __restrict__ Kd,
                                               bf16_t* __restrict__ Vt) {
  const int cn  = lane & 15, hh = lane >> 4;
  const int col = n0 + cn;
  const int sel = col / EMBED;            // 0=q 1=k 2=v
  const int rem = col - sel * EMBED;
  const int head = rem >> 6;
  const int dd   = rem & 63;
#pragma unroll
  for (int r = 0; r < 8; ++r) {
    const int row = m0 + r + 8 * hh;      // token index
    const int b = row >> 10, n = row & (SEQ - 1);
    const size_t hb = (size_t)(b * HEADS + head);
    const float v = c[r];
    if (sel == 0)      Qd[(hb * SEQ + n) * HDIM + dd] = (bf16_t)(v * 0.125f); // 1/sqrt(64)
    else if (sel == 1) Kd[(hb * SEQ + n) * HDIM + dd] = (bf16_t)v;
    else               Vt[(hb * HDIM + dd) * SEQ + n] = (bf16_t)v;
  }
}

__global__ void __launch_bounds__(256) vit_qkv_gemm(const bf16_t* __restrict__ Xb,
                                                    const bf16_t* __restrict__ WT,
                                                    bf16_t* __restrict__ Qd,
                                                    bf16_t* __restrict__ Kd,
                                                    bf16_t* __restrict__ Vt) {
  const int lane = threadIdx.x & 31;
  const int task = blockIdx.x * 8 + (threadIdx.x >> 5);   // one 32x32 tile per wave
  const int NT = THREE_EMBED / 32;                        // 72
  const int tm = task / NT, tn = task - tm * NT;
  const int m0 = tm * 32, n0 = tn * 32;
  const bf16_t* A0 = Xb + (size_t)m0 * EMBED;
  const bf16_t* A1 = Xb + (size_t)(m0 + 16) * EMBED;
  const bf16_t* B0 = WT + (size_t)n0 * EMBED;
  const bf16_t* B1 = WT + (size_t)(n0 + 16) * EMBED;
  v8f c00 = {}, c01 = {}, c10 = {}, c11 = {};
  for (int k0 = 0; k0 < EMBED; k0 += 32) {
    v16bf a0 = load_frag_a(A0, EMBED, lane, k0);
    v16bf a1 = load_frag_a(A1, EMBED, lane, k0);
    v16bf b0 = load_frag_b(B0, EMBED, lane, k0);
    v16bf b1 = load_frag_b(B1, EMBED, lane, k0);
    c00 = wmma_bf16f32(a0, b0, c00);
    c01 = wmma_bf16f32(a0, b1, c01);
    c10 = wmma_bf16f32(a1, b0, c10);
    c11 = wmma_bf16f32(a1, b1, c11);
  }
  store_qkv_tile(c00, m0,      n0,      lane, Qd, Kd, Vt);
  store_qkv_tile(c01, m0,      n0 + 16, lane, Qd, Kd, Vt);
  store_qkv_tile(c10, m0 + 16, n0,      lane, Qd, Kd, Vt);
  store_qkv_tile(c11, m0 + 16, n0 + 16, lane, Qd, Kd, Vt);
}

// ---------------- Flash attention: one 16-query tile per wave ----------------
__global__ void __launch_bounds__(256) vit_flash_attn(const bf16_t* __restrict__ Qd,
                                                      const bf16_t* __restrict__ Kd,
                                                      const bf16_t* __restrict__ Vt,
                                                      bf16_t* __restrict__ Ob) {
  __shared__ alignas(16) bf16_t lds_p[8][16][32];  // per-wave P tile, bf16
  const int lane = threadIdx.x & 31;
  const int wave = threadIdx.x >> 5;
  const int task = blockIdx.x * 8 + wave;          // 0..6143
  const int qt = task & 63;                        // query tile within head
  const int bh = task >> 6;                        // 0..95 (b*12+h)
  const bf16_t* Qh = Qd + ((size_t)bh * SEQ + qt * 16) * HDIM;
  const bf16_t* Kh = Kd + (size_t)bh * SEQ * HDIM;
  const bf16_t* Vh = Vt + (size_t)bh * HDIM * SEQ;  // [feat][key]

  // Q fragments for full d=64 contraction (loaded once)
  const v16bf aq0 = load_frag_a(Qh, HDIM, lane, 0);
  const v16bf aq1 = load_frag_a(Qh, HDIM, lane, 32);

  float mrow[8], lrow[8];
  v8f o0 = {}, o1 = {}, o2 = {}, o3 = {};
#pragma unroll
  for (int r = 0; r < 8; ++r) { mrow[r] = -3.0e38f; lrow[r] = 0.0f; }

  const int hh = lane >> 4, cn = lane & 15;
  for (int j0 = 0; j0 < SEQ; j0 += 32) {
    // S tiles: keys j0..j0+15 (s0) and j0+16..j0+31 (s1); contraction over d=64
    v8f s0 = {}, s1 = {};
    {
      v16bf bk = load_frag_b(Kh + (size_t)j0 * HDIM, HDIM, lane, 0);
      s0 = wmma_bf16f32(aq0, bk, s0);
      bk = load_frag_b(Kh + (size_t)j0 * HDIM, HDIM, lane, 32);
      s0 = wmma_bf16f32(aq1, bk, s0);
      bk = load_frag_b(Kh + (size_t)(j0 + 16) * HDIM, HDIM, lane, 0);
      s1 = wmma_bf16f32(aq0, bk, s1);
      bk = load_frag_b(Kh + (size_t)(j0 + 16) * HDIM, HDIM, lane, 32);
      s1 = wmma_bf16f32(aq1, bk, s1);
    }
    // online softmax; rows live across lanes 0-15 (row r) / 16-31 (row r+8)
#pragma unroll
    for (int r = 0; r < 8; ++r) {
      float t = fmaxf(s0[r], s1[r]);
      t = fmaxf(t, __shfl_xor(t, 1));
      t = fmaxf(t, __shfl_xor(t, 2));
      t = fmaxf(t, __shfl_xor(t, 4));
      t = fmaxf(t, __shfl_xor(t, 8));
      const float mnew  = fmaxf(mrow[r], t);
      const float scale = __expf(mrow[r] - mnew);
      const float p0 = __expf(s0[r] - mnew);
      const float p1 = __expf(s1[r] - mnew);
      s0[r] = p0; s1[r] = p1;
      float rs = p0 + p1;
      rs += __shfl_xor(rs, 1);
      rs += __shfl_xor(rs, 2);
      rs += __shfl_xor(rs, 4);
      rs += __shfl_xor(rs, 8);
      lrow[r] = lrow[r] * scale + rs;
      mrow[r] = mnew;
      o0[r] *= scale; o1[r] *= scale; o2[r] *= scale; o3[r] *= scale;
    }
    // Re-layout P (C-layout) -> A-fragment via LDS bounce
#pragma unroll
    for (int r = 0; r < 8; ++r) {
      lds_p[wave][r + 8 * hh][cn]      = (bf16_t)s0[r];
      lds_p[wave][r + 8 * hh][cn + 16] = (bf16_t)s1[r];
    }
    asm volatile("s_wait_dscnt 0x0" ::: "memory");   // DS in-order per wave
    const v16bf ap = load_frag_a(&lds_p[wave][0][0], 32, lane, 0);
    // O += P(16x32) x V(32keys x 16feat), V from transposed [feat][key] storage
    const v16bf bv0 = load_frag_b(Vh + (size_t)0  * SEQ, SEQ, lane, j0);
    const v16bf bv1 = load_frag_b(Vh + (size_t)16 * SEQ, SEQ, lane, j0);
    const v16bf bv2 = load_frag_b(Vh + (size_t)32 * SEQ, SEQ, lane, j0);
    const v16bf bv3 = load_frag_b(Vh + (size_t)48 * SEQ, SEQ, lane, j0);
    o0 = wmma_bf16f32(ap, bv0, o0);
    o1 = wmma_bf16f32(ap, bv1, o1);
    o2 = wmma_bf16f32(ap, bv2, o2);
    o3 = wmma_bf16f32(ap, bv3, o3);
  }
  // finalize: divide by row sums, store [token][head*64+dd] bf16
  const int b = bh / HEADS, head = bh - b * HEADS;
  const size_t rowbase = (size_t)b * SEQ + qt * 16;
#pragma unroll
  for (int r = 0; r < 8; ++r) {
    const float inv = 1.0f / lrow[r];
    bf16_t* op = Ob + (rowbase + r + 8 * hh) * EMBED + head * HDIM + cn;
    op[0]  = (bf16_t)(o0[r] * inv);
    op[16] = (bf16_t)(o1[r] * inv);
    op[32] = (bf16_t)(o2[r] * inv);
    op[48] = (bf16_t)(o3[r] * inv);
  }
}

// ---------------- Projection GEMM: [8192,768] @ [768,768] + bias -> fp32 out ----------------
__device__ __forceinline__ void store_proj_tile(const v8f& c, int m0, int n0, int lane,
                                                const float* __restrict__ bias,
                                                float* __restrict__ out) {
  const int cn = lane & 15, hh = lane >> 4;
  const int col = n0 + cn;
  const float bv = bias[col];
#pragma unroll
  for (int r = 0; r < 8; ++r)
    out[(size_t)(m0 + r + 8 * hh) * EMBED + col] = c[r] + bv;
}

__global__ void __launch_bounds__(256) vit_proj_gemm(const bf16_t* __restrict__ Ob,
                                                     const bf16_t* __restrict__ WpT,
                                                     const float* __restrict__ bias,
                                                     float* __restrict__ out) {
  const int lane = threadIdx.x & 31;
  const int task = blockIdx.x * 8 + (threadIdx.x >> 5);
  const int NT = EMBED / 32;                       // 24
  const int tm = task / NT, tn = task - tm * NT;
  const int m0 = tm * 32, n0 = tn * 32;
  const bf16_t* A0 = Ob + (size_t)m0 * EMBED;
  const bf16_t* A1 = Ob + (size_t)(m0 + 16) * EMBED;
  const bf16_t* B0 = WpT + (size_t)n0 * EMBED;
  const bf16_t* B1 = WpT + (size_t)(n0 + 16) * EMBED;
  v8f c00 = {}, c01 = {}, c10 = {}, c11 = {};
  for (int k0 = 0; k0 < EMBED; k0 += 32) {
    v16bf a0 = load_frag_a(A0, EMBED, lane, k0);
    v16bf a1 = load_frag_a(A1, EMBED, lane, k0);
    v16bf b0 = load_frag_b(B0, EMBED, lane, k0);
    v16bf b1 = load_frag_b(B1, EMBED, lane, k0);
    c00 = wmma_bf16f32(a0, b0, c00);
    c01 = wmma_bf16f32(a0, b1, c01);
    c10 = wmma_bf16f32(a1, b0, c10);
    c11 = wmma_bf16f32(a1, b1, c11);
  }
  store_proj_tile(c00, m0,      n0,      lane, bias, out);
  store_proj_tile(c01, m0,      n0 + 16, lane, bias, out);
  store_proj_tile(c10, m0 + 16, n0,      lane, bias, out);
  store_proj_tile(c11, m0 + 16, n0 + 16, lane, bias, out);
}

// ---------------- host launch ----------------
extern "C" void kernel_launch(void* const* d_in, const int* in_sizes, int n_in,
                              void* d_out, int out_size, void* d_ws, size_t ws_size,
                              hipStream_t stream) {
  (void)in_sizes; (void)n_in; (void)out_size; (void)ws_size;
  const float* x      = (const float*)d_in[0];   // [8,1024,768]
  const float* W_qkv  = (const float*)d_in[1];   // [768,2304]
  const float* W_proj = (const float*)d_in[2];   // [768,768]
  const float* b_proj = (const float*)d_in[3];   // [768]
  float* out = (float*)d_out;                    // [8,1024,768]

  constexpr size_t XB_N    = (size_t)TOKENS * EMBED;        // 6,291,456
  constexpr size_t WQKVT_N = (size_t)THREE_EMBED * EMBED;   // 1,769,472
  constexpr size_t WPRJT_N = (size_t)EMBED * EMBED;         //   589,824
  constexpr size_t QKV_N   = (size_t)BATCH * HEADS * SEQ * HDIM;  // 6,291,456

  bf16_t* ws = (bf16_t*)d_ws;
  bf16_t* Xb    = ws;              ws += XB_N;
  bf16_t* WqkvT = ws;              ws += WQKVT_N;
  bf16_t* WprjT = ws;              ws += WPRJT_N;
  bf16_t* Qd    = ws;              ws += QKV_N;
  bf16_t* Kd    = ws;              ws += QKV_N;
  bf16_t* Vt    = ws;              ws += QKV_N;
  bf16_t* Ob    = ws;              ws += XB_N;

  // 1) precision conversion / transposes
  vit_cvt_bf16<<<8192, 256, 0, stream>>>(x, Xb, (int)XB_N);
  vit_cvt_transpose<<<4096, 256, 0, stream>>>(W_qkv, WqkvT, EMBED, THREE_EMBED);
  vit_cvt_transpose<<<2304, 256, 0, stream>>>(W_proj, WprjT, EMBED, EMBED);
  // 2) fused QKV GEMM -> Q(scaled)/K/V^T bf16;  (8192/32)*(2304/32)=18432 wave tiles
  vit_qkv_gemm<<<18432 / 8, 256, 0, stream>>>(Xb, WqkvT, Qd, Kd, Vt);
  // 3) flash attention: 8*12*64 = 6144 query tiles
  vit_flash_attn<<<6144 / 8, 256, 0, stream>>>(Qd, Kd, Vt, Ob);
  // 4) projection + bias -> fp32 out; (8192/32)*(768/32)=6144 wave tiles
  vit_proj_gemm<<<6144 / 8, 256, 0, stream>>>(Ob, WprjT, b_proj, out);
}
